// DAGNN_1846835938002
// MI455X (gfx1250) — matrix operand order repaired and checked by reading, not compiled, not run on hardware
//
#include <hip/hip_runtime.h>
#include <hip/hip_bf16.h>

#define NN   100000
#define EE   1600000
#define KDIM 256
#define OUTD 32
#define KHOPS 16

typedef __attribute__((ext_vector_type(16))) __bf16 v16bf;
typedef __attribute__((ext_vector_type(8)))  float  v8f;

__device__ __forceinline__ unsigned short f2bf(float f) {
    __bf16 h = (__bf16)f;                       // RNE convert
    return __builtin_bit_cast(unsigned short, h);
}

// ---------------- f32 -> bf16 bulk convert (vectorized) ----------------
__global__ void cvt_f32_bf16(const float* __restrict__ in,
                             unsigned short* __restrict__ out, long n4) {
    long i = (long)blockIdx.x * blockDim.x + threadIdx.x;
    if (i < n4) {
        float4 v = ((const float4*)in)[i];
        ushort4 o;
        o.x = f2bf(v.x); o.y = f2bf(v.y); o.z = f2bf(v.z); o.w = f2bf(v.w);
        ((ushort4*)out)[i] = o;
    }
}

__global__ void fill_f32(float* __restrict__ p, float v, int n) {
    int i = blockIdx.x * blockDim.x + threadIdx.x;
    if (i < n) p[i] = v;
}

__global__ void deg_kernel(const int* __restrict__ dst, float* __restrict__ degs, int e) {
    int i = blockIdx.x * blockDim.x + threadIdx.x;
    if (i < e) atomicAdd(&degs[dst[i]], 1.0f);
}

__global__ void norm_kernel(const float* __restrict__ degs, float* __restrict__ nrm, int n) {
    int i = blockIdx.x * blockDim.x + threadIdx.x;
    if (i < n) { float d = degs[i]; nrm[i] = (d > 0.0f) ? rsqrtf(d) : 0.0f; }
}

// ---------------- fused bf16 WMMA GEMM:  out = act(A[nrows,256] @ W[256,NOUT] + b) ----------------
// W^T staged into LDS as bf16, +8-element pad (conflict-free ds_load_b128).
// 8 waves/block arranged WAVES_M x WAVES_N; each wave owns MT(=2) row-tiles x NT col-tiles
// (acc = MT*NT*8 VGPRs) so B fragments can be multi-buffered instead of serialized.
template<int NOUT, int NT, bool RELU, bool OUTBF16>
__global__ __launch_bounds__(256)
void mlp_gemm(const unsigned short* __restrict__ A,   // [nrows, 256] bf16
              const float* __restrict__ W,            // [256, NOUT]  f32
              const float* __restrict__ bias,         // [NOUT]
              void* __restrict__ out, int nrows) {
    constexpr int MT       = 2;
    constexpr int WAVES_N  = NOUT / (NT * 16);
    constexpr int WAVES_M  = 8 / WAVES_N;
    constexpr int BLK_ROWS = WAVES_M * MT * 16;
    constexpr int LSTR     = KDIM + 8;                // padded LDS row stride (elements)
    extern __shared__ unsigned short ldsW[];          // [NOUT][LSTR]

    // stage W^T (coalesced global reads, strided LDS writes), convert f32->bf16
    for (int idx = threadIdx.x; idx < NOUT * KDIM; idx += blockDim.x) {
        int n = idx % NOUT;
        int k = idx / NOUT;
        ldsW[n * LSTR + k] = f2bf(W[k * NOUT + n]);
    }
    __syncthreads();

    const int wave  = threadIdx.x >> 5;
    const int lane  = threadIdx.x & 31;
    const int wm    = wave % WAVES_M;
    const int wn    = wave / WAVES_M;
    const int rlane = lane & 15;                      // row-in-tile (A) / col-in-tile (B)
    const int khalf = lane >> 4;
    const int mbase = blockIdx.x * BLK_ROWS + wm * (MT * 16);

    const unsigned short* aptr[MT];
    #pragma unroll
    for (int mt = 0; mt < MT; ++mt) {
        int r = mbase + mt * 16 + rlane;
        if (r >= nrows) r = nrows - 1;
        aptr[mt] = A + (size_t)r * KDIM;
    }

    v8f acc[MT][NT];
    #pragma unroll
    for (int mt = 0; mt < MT; ++mt)
        #pragma unroll
        for (int nt = 0; nt < NT; ++nt)
            #pragma unroll
            for (int j = 0; j < 8; ++j) acc[mt][nt][j] = 0.0f;

    #pragma unroll
    for (int kk = 0; kk < KDIM; kk += 32) {
        // A fragments: K = {kk+8*khalf .. +7} and {kk+16+8*khalf .. +7}
        union { v16bf v; uint4 u[2]; } af[MT];
        const int ab = kk + 8 * khalf;
        #pragma unroll
        for (int mt = 0; mt < MT; ++mt) {
            af[mt].u[0] = *(const uint4*)(aptr[mt] + ab);
            af[mt].u[1] = *(const uint4*)(aptr[mt] + ab + 16);
        }
        // B fragments: col = tile*16+rlane, K = {kk+8*khalf .. +7} and {kk+16+8*khalf .. +7}
        union { v16bf v; uint4 u[2]; } bfr[NT];
        const int bb = kk + 8 * khalf;
        #pragma unroll
        for (int nt = 0; nt < NT; ++nt) {
            const unsigned short* bp = ldsW + ((wn * NT + nt) * 16 + rlane) * LSTR + bb;
            bfr[nt].u[0] = *(const uint4*)(bp);
            bfr[nt].u[1] = *(const uint4*)(bp + 16);
        }
        #pragma unroll
        for (int mt = 0; mt < MT; ++mt)
            #pragma unroll
            for (int nt = 0; nt < NT; ++nt)
                acc[mt][nt] = __builtin_amdgcn_wmma_f32_16x16x32_bf16(
                    false, af[mt].v, false, bfr[nt].v, (short)0, acc[mt][nt], false, false);
    }

    // epilogue: C layout -> VGPR i, lanes<16: row base+i, lanes>=16: row base+8+i, col = lane&15
    const int rowOff = 8 * khalf;
    #pragma unroll
    for (int mt = 0; mt < MT; ++mt) {
        #pragma unroll
        for (int nt = 0; nt < NT; ++nt) {
            int col = (wn * NT + nt) * 16 + rlane;
            float bv = bias[col];
            #pragma unroll
            for (int i = 0; i < 8; ++i) {
                int m = mbase + mt * 16 + rowOff + i;
                if (m < nrows) {
                    float v = acc[mt][nt][i] + bv;
                    if (RELU) v = v > 0.0f ? v : 0.0f;
                    if (OUTBF16) ((unsigned short*)out)[(size_t)m * NOUT + col] = f2bf(v);
                    else         ((float*)out)[(size_t)m * NOUT + col] = v;
                }
            }
        }
    }
}

// ---------------- hop-0 gated pooling init: out = sigmoid(x.s) * x ----------------
__global__ void init_out_kernel(const float* __restrict__ x, const float* __restrict__ s,
                                float* __restrict__ out, int n) {
    int i = blockIdx.x * blockDim.x + threadIdx.x;
    if (i < n) {
        float v[OUTD]; float dot = 0.0f;
        #pragma unroll
        for (int d = 0; d < OUTD; ++d) { v[d] = x[(size_t)i * OUTD + d]; dot += v[d] * s[d]; }
        float g = 1.0f / (1.0f + __expf(-dot));
        #pragma unroll
        for (int d = 0; d < OUTD; ++d) out[(size_t)i * OUTD + d] = g * v[d];
    }
}

// ---------------- per hop: scaled = h*norm ; nxt = 0 ----------------
__global__ void scale_zero_kernel(const float* __restrict__ h, const float* __restrict__ nrm,
                                  float* __restrict__ scaled, float* __restrict__ nxt, int n) {
    int i = blockIdx.x * blockDim.x + threadIdx.x;   // over n*8 float4 chunks
    if (i < n * 8) {
        int node = i >> 3;
        float4 v = ((const float4*)h)[i];
        float m = nrm[node];
        v.x *= m; v.y *= m; v.z *= m; v.w *= m;
        ((float4*)scaled)[i] = v;
        ((float4*)nxt)[i] = make_float4(0.f, 0.f, 0.f, 0.f);
    }
}

// ---------------- per hop: nxt[dst] += scaled[src]  (L2-resident atomics) ----------------
__global__ void scatter_kernel(const float* __restrict__ scaled, const int* __restrict__ src,
                               const int* __restrict__ dst, float* __restrict__ nxt, int e) {
    int i = blockIdx.x * blockDim.x + threadIdx.x;   // over e*8 (4 dims each)
    if (i < e * 8) {
        int eid = i >> 3, g = i & 7;
        int si = src[eid], di = dst[eid];
        float4 v = ((const float4*)(scaled + (size_t)si * OUTD))[g];
        float* d = nxt + (size_t)di * OUTD + g * 4;
        atomicAdd(d + 0, v.x); atomicAdd(d + 1, v.y);
        atomicAdd(d + 2, v.z); atomicAdd(d + 3, v.w);
    }
}

// ---------------- per hop: h = nxt*norm (in place); out += sigmoid(h.s)*h ----------------
__global__ void finalize_kernel(float* __restrict__ h, const float* __restrict__ nrm,
                                const float* __restrict__ s, float* __restrict__ out, int n) {
    int i = blockIdx.x * blockDim.x + threadIdx.x;
    if (i < n) {
        float m = nrm[i];
        float v[OUTD]; float dot = 0.0f;
        #pragma unroll
        for (int d = 0; d < OUTD; ++d) {
            v[d] = h[(size_t)i * OUTD + d] * m;
            dot += v[d] * s[d];
        }
        float g = 1.0f / (1.0f + __expf(-dot));
        #pragma unroll
        for (int d = 0; d < OUTD; ++d) {
            h[(size_t)i * OUTD + d] = v[d];
            out[(size_t)i * OUTD + d] += g * v[d];
        }
    }
}

extern "C" void kernel_launch(void* const* d_in, const int* in_sizes, int n_in,
                              void* d_out, int out_size, void* d_ws, size_t ws_size,
                              hipStream_t stream) {
    (void)in_sizes; (void)n_in; (void)out_size; (void)ws_size;
    const float* feats = (const float*)d_in[0];
    const float* W1 = (const float*)d_in[1];
    const float* b1 = (const float*)d_in[2];
    const float* W2 = (const float*)d_in[3];
    const float* b2 = (const float*)d_in[4];
    const float* W3 = (const float*)d_in[5];
    const float* b3 = (const float*)d_in[6];
    const float* s  = (const float*)d_in[7];
    const int* src  = (const int*)d_in[8];
    const int* dst  = (const int*)d_in[9];
    float* out = (float*)d_out;

    // workspace carve-out (256B aligned)
    char* ws = (char*)d_ws; size_t off = 0;
    auto carve = [&](size_t bytes) { void* p = ws + off; off += (bytes + 255) & ~(size_t)255; return p; };
    unsigned short* bufA = (unsigned short*)carve((size_t)NN * KDIM * 2); // feats_bf16, then h2_bf16
    unsigned short* bufB = (unsigned short*)carve((size_t)NN * KDIM * 2); // h1_bf16
    float* x      = (float*)carve((size_t)NN * OUTD * 4);
    float* hA     = (float*)carve((size_t)NN * OUTD * 4);
    float* hB     = (float*)carve((size_t)NN * OUTD * 4);
    float* scaled = (float*)carve((size_t)NN * OUTD * 4);
    float* degs   = (float*)carve((size_t)NN * 4);
    float* nrm    = (float*)carve((size_t)NN * 4);

    // 1) feats f32 -> bf16
    {
        long n4 = (long)NN * KDIM / 4;
        cvt_f32_bf16<<<(unsigned)((n4 + 255) / 256), 256, 0, stream>>>(feats, bufA, n4);
    }
    // 2) degrees & symmetric norm
    fill_f32<<<(NN + 255) / 256, 256, 0, stream>>>(degs, 0.0f, NN);
    deg_kernel<<<(EE + 255) / 256, 256, 0, stream>>>(dst, degs, EE);
    norm_kernel<<<(NN + 255) / 256, 256, 0, stream>>>(degs, nrm, NN);

    // 3) MLP via bf16 WMMA GEMMs
    const size_t lds256 = (size_t)KDIM * (KDIM + 8) * 2;  // 135168 B
    const size_t lds32  = (size_t)OUTD * (KDIM + 8) * 2;  //  16896 B
    const int gb64  = (NN + 63) / 64;     // 64-row blocks (NOUT=256: 2x4 wave grid)
    const int gb256 = (NN + 255) / 256;   // 256-row blocks (NOUT=32: 8x1 wave grid)
    mlp_gemm<256, 4, true,  true ><<<gb64,  256, lds256, stream>>>(bufA, W1, b1, bufB, NN);
    mlp_gemm<256, 4, true,  true ><<<gb64,  256, lds256, stream>>>(bufB, W2, b2, bufA, NN);
    mlp_gemm< 32, 2, false, false><<<gb256, 256, lds32,  stream>>>(bufA, W3, b3, x,    NN);

    // 4) hop-0 gated pooling
    init_out_kernel<<<(NN + 255) / 256, 256, 0, stream>>>(x, s, out, NN);

    // 5) K hops, fused gating accumulation (ping-pong h buffers; all L2-resident)
    const float* cur = x;
    float* pp[2] = { hA, hB };
    for (int k = 0; k < KHOPS; ++k) {
        float* nxt = pp[k & 1];
        scale_zero_kernel<<<(NN * 8 + 255) / 256, 256, 0, stream>>>(cur, nrm, scaled, nxt, NN);
        scatter_kernel<<<(EE * 8 + 255) / 256, 256, 0, stream>>>(scaled, src, dst, nxt, EE);
        finalize_kernel<<<(NN + 255) / 256, 256, 0, stream>>>(nxt, nrm, s, out, NN);
        cur = nxt;
    }
}
